// GCN_51848845197629
// MI455X (gfx1250) — matrix operand order repaired and checked by reading, not compiled, not run on hardware
//
#include <hip/hip_runtime.h>
#include <hip/hip_bf16.h>

typedef __attribute__((ext_vector_type(16))) _Float16 v16h;
typedef __attribute__((ext_vector_type(8)))  float    v8f;

#define N_NODES  50000
#define N_EDGES  1600000
#define N_GRAPHS 500
#define F3       32

// ---------------- degree / norm ----------------

__global__ void init_deg_kernel(float* deg, int n) {
    int i = blockIdx.x * blockDim.x + threadIdx.x;
    if (i < n) deg[i] = 1.0f;   // self-loop weight 1
}

__global__ void deg_accum_kernel(const long long* __restrict__ dst,
                                 const float* __restrict__ ew,
                                 float* deg, int e) {
    int i = blockIdx.x * blockDim.x + threadIdx.x;
    if (i < e) atomicAdd(&deg[(int)dst[i]], ew[i]);
}

__global__ void dinv_kernel(float* deg, int n) {
    int i = blockIdx.x * blockDim.x + threadIdx.x;
    if (i < n) {
        float d = deg[i];
        deg[i] = (d > 0.0f) ? rsqrtf(d) : 0.0f;   // in-place -> dinv
    }
}

// ---------------- WMMA GEMM: Out[N,Fout] = H[N,Fin] @ W[Fin,Fout] ----------------
// Block = 4 waves, one 16-wide output strip (col0). W strip staged transposed in
// LDS as f16 so each lane's B fragment is a contiguous 32B ds load. Each wave
// computes 4 M-tiles against one resident B fragment (4x WMMA per K-step).
// Tail handling: clamp tile index (redundant waves recompute/store identical
// data) -> branch-free inner loop, EXEC provably all-ones for WMMA.

__global__ void __launch_bounds__(128)
gemm_wmma_kernel(const float* __restrict__ H, const float* __restrict__ W,
                 float* __restrict__ Out, int tilesM, int Fin, int Fout) {
    __shared__ _Float16 Wt[16 * 128];            // [n][k], Fin <= 128 -> <= 4KB

    const int col0 = blockIdx.x * 16;            // output feature base
    // cooperative load + transpose of the W strip (f32 -> f16)
    for (int t = threadIdx.x; t < Fin * 16; t += blockDim.x) {
        int k = t >> 4, n = t & 15;
        Wt[n * Fin + k] = (_Float16)W[(size_t)k * Fout + col0 + n];
    }
    __syncthreads();

    const int wave = threadIdx.x >> 5;
    const int lane = threadIdx.x & 31;
    const int half = lane >> 4;                  // 0: lanes 0-15, 1: lanes 16-31
    const int mr   = lane & 15;                  // A row / B,C col within tile
    const int mtBase = (blockIdx.y * 4 + wave) * 4;   // 4 M-tiles per wave

    // clamp tail tiles: redundant recompute of last tile, identical stores
    int tileM[4];
    #pragma unroll
    for (int mt = 0; mt < 4; ++mt) {
        int t = mtBase + mt;
        tileM[mt] = (t < tilesM) ? t : (tilesM - 1);
    }

    const float* arowb[4];
    #pragma unroll
    for (int mt = 0; mt < 4; ++mt)
        arowb[mt] = H + (size_t)(tileM[mt] * 16 + mr) * Fin;

    v8f acc[4] = {v8f{}, v8f{}, v8f{}, v8f{}};

    for (int k0 = 0; k0 < Fin; k0 += 32) {
        // B fragment: lane col = mr, k = k0 + half*16 + e  -> contiguous 16 f16
        v16h b = *(const v16h*)&Wt[mr * Fin + half * 16 + k0];

        #pragma unroll
        for (int mt = 0; mt < 4; ++mt) {
            const float* arow = arowb[mt] + k0;
            __builtin_prefetch(arow + 32, 0, 3);   // next K-slab
            // A fragment: k = half*16 + [0..15]
            float4 a0 = *(const float4*)(arow + half * 16);
            float4 a1 = *(const float4*)(arow + half * 16 + 4);
            float4 a2 = *(const float4*)(arow + half * 16 + 8);
            float4 a3 = *(const float4*)(arow + half * 16 + 12);
            float af[16] = {a0.x, a0.y, a0.z, a0.w, a1.x, a1.y, a1.z, a1.w,
                            a2.x, a2.y, a2.z, a2.w, a3.x, a3.y, a3.z, a3.w};
            v16h a;
            #pragma unroll
            for (int e = 0; e < 16; ++e) a[e] = (_Float16)af[e];
            acc[mt] = __builtin_amdgcn_wmma_f32_16x16x32_f16(
                false, a, false, b, (short)0, acc[mt], false, false);
        }
    }

    // C/D layout: VGPR r -> M = r + half*8, N = mr
    #pragma unroll
    for (int mt = 0; mt < 4; ++mt) {
        float* o = Out + (size_t)(tileM[mt] * 16) * Fout + col0 + mr;
        #pragma unroll
        for (int r = 0; r < 8; ++r)
            o[(size_t)(r + half * 8) * Fout] = acc[mt][r];
    }
}

// ---------------- aggregation ----------------
// Hagg[n,f] = bias[f] + dinv[n]^2 * Hw[n,f]   (self loop, norm = dinv*1*dinv)

__global__ void agg_init_kernel(const float* __restrict__ Hw,
                                const float* __restrict__ dinv,
                                const float* __restrict__ bias,
                                float* __restrict__ Hagg, int n, int F) {
    long long i = (long long)blockIdx.x * blockDim.x + threadIdx.x;
    long long tot = (long long)n * F;
    if (i >= tot) return;
    int node = (int)(i / F), f = (int)(i % F);
    float di = dinv[node];
    Hagg[i] = bias[f] + di * di * Hw[i];
}

// per-edge scatter: 4 features per thread, float atomics land in L2 (graph fits in 192MB L2)
__global__ void edge_scatter_kernel(const float* __restrict__ Hw,
                                    const long long* __restrict__ src,
                                    const long long* __restrict__ dst,
                                    const float* __restrict__ ew,
                                    const float* __restrict__ dinv,
                                    float* __restrict__ Hagg, int e, int F) {
    long long tid = (long long)blockIdx.x * blockDim.x + threadIdx.x;
    int chunks = F >> 2;
    long long tot = (long long)e * chunks;
    if (tid >= tot) return;
    int ei = (int)(tid / chunks);
    int c  = (int)(tid % chunks);
    long long s = src[ei], d = dst[ei];
    float coef = dinv[(int)s] * ew[ei] * dinv[(int)d];
    const float4 v = *(const float4*)(Hw + s * F + (size_t)c * 4);
    float* o = Hagg + d * F + (size_t)c * 4;
    atomicAdd(o + 0, coef * v.x);
    atomicAdd(o + 1, coef * v.y);
    atomicAdd(o + 2, coef * v.z);
    atomicAdd(o + 3, coef * v.w);
}

// ---------------- conv1d over feature axis, fused ReLU in + ReLU out ----------------

__global__ void conv_relu_kernel(const float* __restrict__ Hagg,
                                 const float* __restrict__ kw,
                                 const float* __restrict__ cb,
                                 float* __restrict__ Out,
                                 int n, int F, int ks, int pad) {
    long long i = (long long)blockIdx.x * blockDim.x + threadIdx.x;
    long long tot = (long long)n * F;
    if (i >= tot) return;
    int node = (int)(i / F), f = (int)(i % F);
    const float* row = Hagg + (size_t)node * F;
    float acc = cb[0];
    for (int j = 0; j < ks; ++j) {
        int ff = f + j - pad;
        if (ff >= 0 && ff < F) {
            float v = row[ff];
            v = v > 0.0f ? v : 0.0f;        // ReLU on gcn output
            acc += kw[j] * v;
        }
    }
    Out[i] = acc > 0.0f ? acc : 0.0f;       // ReLU on conv output
}

// ---------------- global mean pool ----------------

__global__ void pool_zero_kernel(float* pool, float* cnt) {
    int i = blockIdx.x * blockDim.x + threadIdx.x;
    if (i < N_GRAPHS * F3) pool[i] = 0.0f;
    if (i < N_GRAPHS) cnt[i] = 0.0f;
}

__global__ void pool_accum_kernel(const float* __restrict__ H,
                                  const long long* __restrict__ batch,
                                  float* pool, float* cnt, int n, int F) {
    long long i = (long long)blockIdx.x * blockDim.x + threadIdx.x;
    long long tot = (long long)n * F;
    if (i >= tot) return;
    int node = (int)(i / F), f = (int)(i % F);
    int g = (int)batch[node];
    atomicAdd(&pool[(size_t)g * F + f], H[i]);
    if (f == 0) atomicAdd(&cnt[g], 1.0f);
}

__global__ void pool_final_kernel(const float* pool, const float* cnt,
                                  float* out, int G, int F) {
    int i = blockIdx.x * blockDim.x + threadIdx.x;
    if (i >= G * F) return;
    int g = i / F;
    float c = cnt[g];
    out[i] = pool[i] / (c > 1.0f ? c : 1.0f);
}

// ---------------- driver ----------------

static inline int cdiv(long long a, long long b) { return (int)((a + b - 1) / b); }

extern "C" void kernel_launch(void* const* d_in, const int* in_sizes, int n_in,
                              void* d_out, int out_size, void* d_ws, size_t ws_size,
                              hipStream_t stream) {
    const float*     x     = (const float*)d_in[0];
    const long long* eidx  = (const long long*)d_in[1];   // int64 in reference
    const long long* esrc  = eidx;
    const long long* edst  = eidx + N_EDGES;
    const float*     ew    = (const float*)d_in[2];
    const long long* batch = (const long long*)d_in[3];
    const float* W1 = (const float*)d_in[4];
    const float* b1 = (const float*)d_in[5];
    const float* k1 = (const float*)d_in[6];
    const float* cb1 = (const float*)d_in[7];
    const float* W2 = (const float*)d_in[8];
    const float* b2 = (const float*)d_in[9];
    const float* k2 = (const float*)d_in[10];
    const float* cb2 = (const float*)d_in[11];
    const float* W3 = (const float*)d_in[12];
    const float* b3 = (const float*)d_in[13];
    const float* k3 = (const float*)d_in[14];
    const float* cb3 = (const float*)d_in[15];
    float* out = (float*)d_out;

    // workspace layout
    const size_t bufBytes = (size_t)N_NODES * 128 * sizeof(float);   // 25.6 MB
    char* ws = (char*)d_ws;
    float* dinv = (float*)ws;                         // N floats (deg -> dinv in place)
    float* cnt  = dinv + N_NODES;                     // N_GRAPHS floats
    float* pool = cnt + N_GRAPHS;                     // N_GRAPHS*F3 floats
    float* bufA = (float*)(ws + (1 << 20));
    float* bufB = (float*)(ws + (1 << 20) + bufBytes);
    if (ws_size < (1 << 20) + 2 * bufBytes) return;   // defensive; deterministic

    const int B = 256;
    const int tilesM = N_NODES / 16;                  // 3125, exact

    // gcn_norm
    init_deg_kernel<<<cdiv(N_NODES, B), B, 0, stream>>>(dinv, N_NODES);
    deg_accum_kernel<<<cdiv(N_EDGES, B), B, 0, stream>>>(edst, ew, dinv, N_EDGES);
    dinv_kernel<<<cdiv(N_NODES, B), B, 0, stream>>>(dinv, N_NODES);

    struct Layer {
        const float *in, *W, *b, *k, *cb; float *hw, *hagg, *outb;
        int Fin, Fout, ks, pad;
    };
    Layer L[3] = {
        { x,    W1, b1, k1, cb1, bufA, bufB, bufA, 128, 128, 5, 2 },
        { bufA, W2, b2, k2, cb2, bufB, bufA, bufB, 128,  64, 5, 2 },
        { bufB, W3, b3, k3, cb3, bufA, bufB, bufA,  64,  32, 3, 1 },
    };

    for (int l = 0; l < 3; ++l) {
        const int Fout = L[l].Fout;
        // GEMM (WMMA): grid.x = feature strips, grid.y = M-tile groups (16/block)
        dim3 ggrid(Fout >> 4, cdiv(tilesM, 16));
        gemm_wmma_kernel<<<ggrid, 128, 0, stream>>>(
            L[l].in, L[l].W, L[l].hw, tilesM, L[l].Fin, Fout);
        // self loop + bias
        agg_init_kernel<<<cdiv((long long)N_NODES * Fout, B), B, 0, stream>>>(
            L[l].hw, dinv, L[l].b, L[l].hagg, N_NODES, Fout);
        // edge scatter (L2-resident atomics)
        edge_scatter_kernel<<<cdiv((long long)N_EDGES * (Fout >> 2), B), B, 0, stream>>>(
            L[l].hw, esrc, edst, ew, dinv, L[l].hagg, N_EDGES, Fout);
        // conv1d + fused ReLUs
        conv_relu_kernel<<<cdiv((long long)N_NODES * Fout, B), B, 0, stream>>>(
            L[l].hagg, L[l].k, L[l].cb, L[l].outb, N_NODES, Fout, L[l].ks, L[l].pad);
    }

    // global mean pool (final features live in bufA)
    pool_zero_kernel<<<cdiv(N_GRAPHS * F3, B), B, 0, stream>>>(pool, cnt);
    pool_accum_kernel<<<cdiv((long long)N_NODES * F3, B), B, 0, stream>>>(
        bufA, batch, pool, cnt, N_NODES, F3);
    pool_final_kernel<<<cdiv(N_GRAPHS * F3, B), B, 0, stream>>>(
        pool, cnt, out, N_GRAPHS, F3);
}
